// Node2AnchorAttention_56023553409073
// MI455X (gfx1250) — matrix operand order repaired and checked by reading, not compiled, not run on hardware
//
#include <hip/hip_runtime.h>
#include <hip/hip_bf16.h>
#include <math.h>

// ---------------------------------------------------------------------------
// Node->Anchor graph attention for MI455X (gfx1250, wave32, WMMA).
// Heavy GEMM (500k x 192 x 256 ~ 24.6 GMAC) via v_wmma_f32_16x16x32_f16.
// f16 weights are staged global->LDS with GLOBAL_LOAD_ASYNC_TO_LDS_B128 once
// per block and reused for 16 edge-tiles; A tiles are gathered+converted into
// per-wave LDS. node_features / anchor_q stay L2-resident (192MB L2), so the
// V-half is recomputed in a 2nd pass instead of spilling 256MB of kv.
// ---------------------------------------------------------------------------

#define H_NODE   128
#define H_EDGE   64
#define H_IN     192
#define N_ANCH   20000
#define N_EDGES  500000
#define E_TILES  31250      // 500000 / 16
#define A_TILES  1250       // 20000 / 16
#define KS       200        // LDS row stride in halfwords (padded: 16-row frag reads conflict-free)

typedef __attribute__((ext_vector_type(16))) _Float16     v16h;
typedef __attribute__((ext_vector_type(2)))  __fp16       fp16x2;
typedef __attribute__((ext_vector_type(8)))  float        v8f;
typedef __attribute__((ext_vector_type(4)))  float        f32x4;
typedef __attribute__((ext_vector_type(4)))  unsigned int u32x4;

union Frag16 { v16h h; u32x4 u[2]; };

// ---- workspace layout (bytes), ~12.54 MB total -----------------------------
#define WS_QW16    0                 // 128*128 f16  = 32768 B
#define WS_KVW16   32768             // 256*192 f16  = 98304 B
#define WS_AQ      131072            // 20000*128 f32 = 10,240,000 B
#define WS_LOGIT   10371072          // 500000 f32   = 2,000,000 B
#define WS_SEGMAX  12371072          // 20000 f32
#define WS_SEGSUM  12451072          // 20000 f32

static __device__ __forceinline__ unsigned int pk2h(float a, float b) {
    fp16x2 p = __builtin_amdgcn_cvt_pkrtz(a, b);   // v_cvt_pk_rtz_f16_f32
    return __builtin_bit_cast(unsigned int, p);
}

static __device__ __forceinline__ unsigned short f2h(float a) {
    _Float16 ha = (_Float16)a;
    return __builtin_bit_cast(unsigned short, ha);
}

// A-fragment (16x32 f16, MxK): lane row M = lane&15,
// half = lane>>4; VGPR0..7 = K(half*16 + 0..15) (matches B-fragment order)
static __device__ __forceinline__ v16h load_afrag(const unsigned short* sm,
                                                  int row, int kbase, int half) {
    Frag16 f;
    const int o0 = row * KS + kbase + half * 16;
    f.u[0] = *(const u32x4*)(sm + o0);
    f.u[1] = *(const u32x4*)(sm + o0 + 8);
    return f.h;
}

// B-fragment (32x16 f16, KxN): lane = column N; 16 consecutive f16 along K.
static __device__ __forceinline__ v16h load_bfrag(const unsigned short* p, int off) {
    Frag16 f;
    f.u[0] = *(const u32x4*)(p + off);
    f.u[1] = *(const u32x4*)(p + off + 8);
    return f.h;
}

static __device__ __forceinline__ void atomicMaxF(float* addr, float v) {
    if (v >= 0.0f) atomicMax((int*)addr, __float_as_int(v));
    else           atomicMin((unsigned int*)addr, __float_as_uint(v));
}

// Stage 128 weight rows (rowHW halfwords each, 16B-chunked) from global into
// LDS (stride KS) using the CDNA5 async DMA path. Wave w covers rows w*64..+64.
static __device__ __forceinline__ void stage_weights_async(
        const unsigned short* __restrict__ g, int rowHW,
        unsigned short* smB, int lane, int wave) {
    const int cpr   = rowHW >> 3;          // 16B chunks per row
    const int iters = (64 * cpr) >> 5;     // chunks per wave / 32 lanes
    #pragma unroll 4
    for (int i = 0; i < iters; ++i) {
        const int c   = i * 32 + lane;
        const int row = (wave << 6) + c / cpr;
        const int cir = c % cpr;
        unsigned long long ga =
            (unsigned long long)(const void*)(g + (long)row * rowHW + (cir << 3));
        unsigned int la =
            (unsigned int)(unsigned long long)(const void*)(smB + row * KS + (cir << 3));
        asm volatile("global_load_async_to_lds_b128 %0, %1, off"
                     :: "v"(la), "v"(ga) : "memory");
    }
    asm volatile("s_wait_asynccnt 0x0" ::: "memory");
}

// ---------------------------------------------------------------------------
// K0: init output/segment buffers, convert weights to f16
// ---------------------------------------------------------------------------
__global__ void k0_init(float* out, float* segmax, float* segsum,
                        const float* qw, unsigned short* qw16,
                        const float* kvw, unsigned short* kvw16) {
    const int i = blockIdx.x * blockDim.x + threadIdx.x;
    if (i < N_ANCH * H_NODE) out[i] = 0.0f;
    if (i < N_ANCH) { segmax[i] = -INFINITY; segsum[i] = 0.0f; }
    if (i < H_NODE * H_NODE) qw16[i] = f2h(qw[i]);
    if (i < 2 * H_NODE * H_IN) kvw16[i] = f2h(kvw[i]);
}

// ---------------------------------------------------------------------------
// K1: anchor_q = anchor_features @ q_w^T + q_b
// ---------------------------------------------------------------------------
__global__ void __launch_bounds__(64)
k1_anchor_q(const float* __restrict__ anchorf,
            const unsigned short* __restrict__ qw16,
            const float* __restrict__ qb,
            float* __restrict__ anchor_q) {
    __shared__ unsigned short smB[128 * KS];
    __shared__ unsigned short smA[2 * 16 * KS];
    const int wave = threadIdx.x >> 5;
    const int lane = threadIdx.x & 31;
    stage_weights_async(qw16, H_NODE, smB, lane, wave);
    __syncthreads();

    unsigned short* smw = smA + wave * 16 * KS;
    const int half = lane >> 4, nlo = lane & 15;

    for (int it = 0; it < 8; ++it) {
        const int tile = blockIdx.x * 16 + wave * 8 + it;
        const bool act = (tile < A_TILES);
        const int tbase = tile * 16;

        if (act) {  // fill A tile: 16 anchors x 128 cols, f32 -> f16
            const int e = lane >> 1, sub = lane & 1;
            unsigned int* rowp = (unsigned int*)(smw + e * KS);
            const float* src = anchorf + (long)(tbase + e) * H_NODE;
            #pragma unroll
            for (int c = 0; c < 64; c += 4) {
                const int col = sub * 64 + c;
                f32x4 x = *(const f32x4*)(src + col);
                rowp[(col >> 1) + 0] = pk2h(x[0], x[1]);
                rowp[(col >> 1) + 1] = pk2h(x[2], x[3]);
            }
        }
        __syncthreads();
        if (act) {
            v8f acc[8];
            #pragma unroll
            for (int nt = 0; nt < 8; ++nt) acc[nt] = {};
            #pragma unroll
            for (int kc = 0; kc < 4; ++kc) {
                v16h a = load_afrag(smw, nlo, kc * 32, half);
                #pragma unroll
                for (int nt = 0; nt < 8; ++nt) {
                    v16h b = load_bfrag(smB, (nt * 16 + nlo) * KS + kc * 32 + half * 16);
                    acc[nt] = __builtin_amdgcn_wmma_f32_16x16x32_f16(
                        false, a, false, b, (short)0, acc[nt], false, false);
                }
            }
            #pragma unroll
            for (int nt = 0; nt < 8; ++nt) {
                const int ncol = nt * 16 + nlo;
                const float bias = qb[ncol];
                #pragma unroll
                for (int r = 0; r < 8; ++r)
                    anchor_q[(long)(tbase + r + 8 * half) * H_NODE + ncol] = acc[nt][r] + bias;
            }
        }
        __syncthreads();
    }
}

// fill 16 edges x 192 cols (node_features[src] || edge_features), f32 -> f16
static __device__ __forceinline__ void fill_edge_tile(unsigned short* smw, int lane,
                                                      int tbase,
                                                      const int* __restrict__ srcIdx,
                                                      const float* __restrict__ nodef,
                                                      const float* __restrict__ edgef) {
    const int e = lane >> 1, sub = lane & 1;
    const int eg = tbase + e;
    const int sn = srcIdx[eg];
    unsigned int* rowp = (unsigned int*)(smw + e * KS);
    const float* nrow = nodef + (long)sn * H_NODE;
    const float* erow = edgef + (long)eg * H_EDGE;
    #pragma unroll
    for (int c = 0; c < 96; c += 4) {
        const int col = sub * 96 + c;
        f32x4 x = (col < H_NODE) ? *(const f32x4*)(nrow + col)
                                 : *(const f32x4*)(erow + (col - H_NODE));
        rowp[(col >> 1) + 0] = pk2h(x[0], x[1]);
        rowp[(col >> 1) + 1] = pk2h(x[2], x[3]);
    }
}

// ---------------------------------------------------------------------------
// K2: logits = q[dst] . (concat @ kv_w[:128]^T + kv_b[:128]); atomicMax segmax
// ---------------------------------------------------------------------------
__global__ void __launch_bounds__(64)
k2_logits(const float* __restrict__ nodef, const float* __restrict__ edgef,
          const int* __restrict__ eidx, const float* __restrict__ kvb,
          const unsigned short* __restrict__ kvw16,
          const float* __restrict__ anchor_q,
          float* __restrict__ logit, float* __restrict__ segmax) {
    __shared__ unsigned short smB[128 * KS];
    __shared__ unsigned short smA[2 * 16 * KS];
    const int wave = threadIdx.x >> 5;
    const int lane = threadIdx.x & 31;
    stage_weights_async(kvw16, H_IN, smB, lane, wave);   // K-half rows 0..127
    __syncthreads();

    const int* srcIdx = eidx;
    const int* dstIdx = eidx + N_EDGES;
    unsigned short* smw = smA + wave * 16 * KS;
    const int half = lane >> 4, nlo = lane & 15;

    for (int it = 0; it < 8; ++it) {
        const int tile = blockIdx.x * 16 + wave * 8 + it;
        const bool act = (tile < E_TILES);
        const int tbase = tile * 16;

        if (act) fill_edge_tile(smw, lane, tbase, srcIdx, nodef, edgef);
        const int ntile = tile + 1;                       // prefetch next edge rows
        if (it < 7 && ntile < E_TILES)
            __builtin_prefetch(edgef + (long)ntile * 16 * H_EDGE + lane * 32, 0, 1);
        __syncthreads();

        if (act) {
            int dstM[8];
            #pragma unroll
            for (int r = 0; r < 8; ++r) dstM[r] = dstIdx[tbase + r + 8 * half];

            v8f acc[8];
            #pragma unroll
            for (int nt = 0; nt < 8; ++nt) acc[nt] = {};
            #pragma unroll
            for (int kc = 0; kc < 6; ++kc) {
                v16h a = load_afrag(smw, nlo, kc * 32, half);
                #pragma unroll
                for (int nt = 0; nt < 8; ++nt) {
                    v16h b = load_bfrag(smB, (nt * 16 + nlo) * KS + kc * 32 + half * 16);
                    acc[nt] = __builtin_amdgcn_wmma_f32_16x16x32_f16(
                        false, a, false, b, (short)0, acc[nt], false, false);
                }
            }
            float partial[8] = {0, 0, 0, 0, 0, 0, 0, 0};
            #pragma unroll
            for (int nt = 0; nt < 8; ++nt) {
                const int ncol = nt * 16 + nlo;
                const float bias = kvb[ncol];
                #pragma unroll
                for (int r = 0; r < 8; ++r) {
                    // coalesced: lanes 0..15 read 16 contiguous floats of q[dst]
                    const float qv = anchor_q[(long)dstM[r] * H_NODE + ncol];
                    partial[r] += (acc[nt][r] + bias) * qv;
                }
            }
            #pragma unroll
            for (int r = 0; r < 8; ++r) {
                float v = partial[r];
                v += __shfl_xor(v, 1, 16);
                v += __shfl_xor(v, 2, 16);
                v += __shfl_xor(v, 4, 16);
                v += __shfl_xor(v, 8, 16);
                if (nlo == 0) {
                    logit[tbase + r + 8 * half] = v;
                    atomicMaxF(segmax + dstM[r], v);
                }
            }
        }
        __syncthreads();
    }
}

// ---------------------------------------------------------------------------
// K3: w = exp(logit - segmax[dst]); segsum += w; out[dst] += w * v  (v half)
// ---------------------------------------------------------------------------
__global__ void __launch_bounds__(64)
k3_value(const float* __restrict__ nodef, const float* __restrict__ edgef,
         const int* __restrict__ eidx, const float* __restrict__ kvb,
         const unsigned short* __restrict__ kvw16,
         const float* __restrict__ logit, const float* __restrict__ segmax,
         float* __restrict__ segsum, float* __restrict__ out) {
    __shared__ unsigned short smB[128 * KS];
    __shared__ unsigned short smA[2 * 16 * KS];
    const int wave = threadIdx.x >> 5;
    const int lane = threadIdx.x & 31;
    stage_weights_async(kvw16 + (long)H_NODE * H_IN, H_IN, smB, lane, wave); // V-half
    __syncthreads();

    const int* srcIdx = eidx;
    const int* dstIdx = eidx + N_EDGES;
    unsigned short* smw = smA + wave * 16 * KS;
    const int half = lane >> 4, nlo = lane & 15;

    for (int it = 0; it < 8; ++it) {
        const int tile = blockIdx.x * 16 + wave * 8 + it;
        const bool act = (tile < E_TILES);
        const int tbase = tile * 16;

        if (act) fill_edge_tile(smw, lane, tbase, srcIdx, nodef, edgef);
        const int ntile = tile + 1;
        if (it < 7 && ntile < E_TILES)
            __builtin_prefetch(edgef + (long)ntile * 16 * H_EDGE + lane * 32, 0, 1);
        __syncthreads();

        if (act) {
            int dstM[8];
            float w[8];
            #pragma unroll
            for (int r = 0; r < 8; ++r) {
                const int e = tbase + r + 8 * half;
                dstM[r] = dstIdx[e];
                w[r] = __expf(logit[e] - segmax[dstM[r]]);
            }
            if (nlo == 0) {
                #pragma unroll
                for (int r = 0; r < 8; ++r) atomicAdd(segsum + dstM[r], w[r]);
            }
            v8f acc[8];
            #pragma unroll
            for (int nt = 0; nt < 8; ++nt) acc[nt] = {};
            #pragma unroll
            for (int kc = 0; kc < 6; ++kc) {
                v16h a = load_afrag(smw, nlo, kc * 32, half);
                #pragma unroll
                for (int nt = 0; nt < 8; ++nt) {
                    v16h b = load_bfrag(smB, (nt * 16 + nlo) * KS + kc * 32 + half * 16);
                    acc[nt] = __builtin_amdgcn_wmma_f32_16x16x32_f16(
                        false, a, false, b, (short)0, acc[nt], false, false);
                }
            }
            #pragma unroll
            for (int nt = 0; nt < 8; ++nt) {
                const int ncol = nt * 16 + nlo;
                const float bias = kvb[H_NODE + ncol];
                #pragma unroll
                for (int r = 0; r < 8; ++r)   // cacheline-coalesced f32 atomics
                    atomicAdd(out + (long)dstM[r] * H_NODE + ncol, w[r] * (acc[nt][r] + bias));
            }
        }
        __syncthreads();
    }
}

// ---------------------------------------------------------------------------
// K4: out /= (segsum + 1e-16)
// ---------------------------------------------------------------------------
__global__ void k4_norm(float* __restrict__ out, const float* __restrict__ segsum) {
    const int i = blockIdx.x * blockDim.x + threadIdx.x;
    if (i < N_ANCH * H_NODE)
        out[i] = out[i] / (segsum[i >> 7] + 1e-16f);
}

// ---------------------------------------------------------------------------
extern "C" void kernel_launch(void* const* d_in, const int* in_sizes, int n_in,
                              void* d_out, int out_size, void* d_ws, size_t ws_size,
                              hipStream_t stream) {
    const float* anchorf = (const float*)d_in[0];   // (20000,128)
    const float* nodef   = (const float*)d_in[1];   // (20000,128)
    const int*   eidx    = (const int*)d_in[2];     // (2,500000)
    const float* edgef   = (const float*)d_in[3];   // (500000,64)
    const float* qw      = (const float*)d_in[4];   // (128,128)
    const float* qb      = (const float*)d_in[5];   // (128)
    const float* kvw     = (const float*)d_in[6];   // (256,192)
    const float* kvb     = (const float*)d_in[7];   // (256)
    float* out = (float*)d_out;

    char* ws = (char*)d_ws;
    unsigned short* qw16   = (unsigned short*)(ws + WS_QW16);
    unsigned short* kvw16  = (unsigned short*)(ws + WS_KVW16);
    float*          aq     = (float*)(ws + WS_AQ);
    float*          logit  = (float*)(ws + WS_LOGIT);
    float*          segmax = (float*)(ws + WS_SEGMAX);
    float*          segsum = (float*)(ws + WS_SEGSUM);

    k0_init<<<(N_ANCH * H_NODE + 255) / 256, 256, 0, stream>>>(
        out, segmax, segsum, qw, qw16, kvw, kvw16);

    k1_anchor_q<<<(A_TILES + 15) / 16, 64, 0, stream>>>(anchorf, qw16, qb, aq);

    k2_logits<<<(E_TILES + 15) / 16, 64, 0, stream>>>(
        nodef, edgef, eidx, kvb, kvw16, aq, logit, segmax);

    k3_value<<<(E_TILES + 15) / 16, 64, 0, stream>>>(
        nodef, edgef, eidx, kvb, kvw16, logit, segmax, segsum, out);

    k4_norm<<<(N_ANCH * H_NODE + 255) / 256, 256, 0, stream>>>(out, segsum);
}
